// DotProductAttention_6030134084023
// MI455X (gfx1250) — matrix operand order repaired and checked by reading, not compiled, not run on hardware
//
#include <hip/hip_runtime.h>
#include <type_traits>

// ---------------------------------------------------------------------------
// Flash attention for B=64, S=1024, D=64 (fp32 in/out), bf16 WMMA compute,
// async global->LDS K staging, software-pipelined (double-buffered) tiles,
// DPP row_xmask softmax reductions, masked tile peeled out of the hot loop.
// ---------------------------------------------------------------------------

#define NEGV (-1000000.0f)

typedef __attribute__((ext_vector_type(16))) __bf16 v16bf;
typedef __attribute__((ext_vector_type(8)))  float  v8f;
typedef __attribute__((ext_vector_type(4)))  int    v4i;

constexpr int BATCH = 64;
constexpr int SEQ   = 1024;
constexpr int DIM   = 64;
constexpr int QT    = 128;  // query rows per block (8 waves x 16 rows)
constexpr int KT    = 64;   // keys per tile
constexpr int PITCH = 72;   // bf16 row pitch in LDS (multiple of 8 -> 16B rows)

union FragBF { uint4 u[2]; v16bf v; };

__device__ inline v16bf make_frag(uint4 a, uint4 b) {
  FragBF f; f.u[0] = a; f.u[1] = b; return f.v;
}

__device__ inline v8f zero_v8() {
  v8f z;
#pragma unroll
  for (int i = 0; i < 8; ++i) z[i] = 0.0f;
  return z;
}

// ---- 16-lane-row xor butterfly step (DPP ROW_XMASK; VALU, no LDS traffic) --
template <int MASK>
__device__ inline float red_step(float x) {
#if __has_builtin(__builtin_amdgcn_mov_dpp) && __has_builtin(__builtin_bit_cast)
  int i = __builtin_bit_cast(int, x);
  i = __builtin_amdgcn_mov_dpp(i, 0x160 | MASK, 0xf, 0xf, true);  // ROW_XMASK:MASK
  return __builtin_bit_cast(float, i);
#else
  return __shfl_xor(x, MASK, 32);
#endif
}

__device__ inline float rowmax16(float x) {
  x = fmaxf(x, red_step<1>(x));
  x = fmaxf(x, red_step<2>(x));
  x = fmaxf(x, red_step<4>(x));
  x = fmaxf(x, red_step<8>(x));
  return x;
}

__device__ inline float rowsum16(float x) {
  x += red_step<1>(x);
  x += red_step<2>(x);
  x += red_step<4>(x);
  x += red_step<8>(x);
  return x;
}

// ---------------- async global->LDS staging (CDNA5 path, guarded) ----------
#if defined(__gfx1250__) && __has_builtin(__builtin_amdgcn_global_load_async_to_lds_b128)
typedef __attribute__((address_space(1))) v4i gv4i;  // global int4
typedef __attribute__((address_space(3))) v4i lv4i;  // LDS int4
__device__ inline void stage16(void* lds, const void* g) {
  __builtin_amdgcn_global_load_async_to_lds_b128((gv4i*)g, (lv4i*)lds, 0, 0);
}
__device__ inline void stage_wait() {
#if __has_builtin(__builtin_amdgcn_s_wait_asynccnt)
  __builtin_amdgcn_s_wait_asynccnt(0);
#else
  asm volatile("s_wait_asynccnt 0x0" ::: "memory");
#endif
}
#else
__device__ inline void stage16(void* lds, const void* g) {
  *(uint4*)lds = *(const uint4*)g;
}
__device__ inline void stage_wait() {}
#endif

// ---------------------------------------------------------------------------
__global__ __launch_bounds__(256, 1)
void DotProductAttention_flash_bf16_wmma(const float* __restrict__ Q,
                                         const float* __restrict__ K,
                                         const float* __restrict__ V,
                                         const int*   __restrict__ VL,
                                         float* __restrict__ Out) {
  // raw async landing zone for K (single buffer: each thread converts only the
  // chunks it staged itself, so only its own ASYNCcnt ordering matters)
  __shared__ __attribute__((aligned(16))) float  sKf[KT * DIM];
  // double-buffered bf16 operand tiles
  __shared__ __attribute__((aligned(16))) __bf16 sKb[2][KT * PITCH];   // [key][d]
  __shared__ __attribute__((aligned(16))) __bf16 sVT[2][DIM * PITCH];  // [d][key]
  // per-wave P scratch (C-layout -> A-layout relayout)
  __shared__ __attribute__((aligned(16))) __bf16 sP[8 * 16 * PITCH];

  const int tid  = threadIdx.x;
  const int wave = tid >> 5;
  const int lane = tid & 31;
  const int m16  = lane & 15;   // column / row-within-16 index
  const int half = lane >> 4;   // which 16-lane half

  const int b     = blockIdx.x >> 3;            // SEQ/QT == 8 q-tiles per batch
  const int qtile = blockIdx.x & 7;
  const int qbase = qtile * QT + wave * 16;     // this wave's 16 query rows
  const int valid = VL[b];

  const float* Kb_ = K + (size_t)b * SEQ * DIM;
  const float* Vb_ = V + (size_t)b * SEQ * DIM;
  const int  nfull      = valid >> 6;           // fully-valid tiles
  const bool haspartial = (valid & (KT - 1)) != 0;
  const int  ktiles     = nfull + (haspartial ? 1 : 0);

  // ---- Q fragments (A-matrix layout, 2 K-chunks of 32), pre-scaled by 1/sqrt(D)
  v16bf qa[2];
  {
    const float* qrow = Q + ((size_t)(b * SEQ + qbase + m16)) * DIM;
#pragma unroll
    for (int c = 0; c < 2; ++c) {
      const int k0 = 32 * c + 8 * half;       // run 1: 8 elems
      const int k1 = 32 * c + 16 + 8 * half;  // run 2: 8 elems
      FragBF f;
      __bf16* bp = (__bf16*)&f;
#pragma unroll
      for (int j = 0; j < 8; ++j) bp[j]     = (__bf16)(qrow[k0 + j] * 0.125f);
#pragma unroll
      for (int j = 0; j < 8; ++j) bp[8 + j] = (__bf16)(qrow[k1 + j] * 0.125f);
      qa[c] = f.v;
    }
  }

  // ---- staging helpers (software pipeline) --------------------------------
  float4 vreg[4];  // in-flight V tile (16 f32 per thread, coalesced)

  auto stage_issue = [&](int kt) {
    const float* srcK = Kb_ + (size_t)kt * KT * DIM;
    const float* srcV = Vb_ + (size_t)kt * KT * DIM;
#pragma unroll
    for (int i = 0; i < 4; ++i) {
      const int chunk = tid + i * 256;          // 16-byte chunk id (1024 total)
      stage16(&sKf[chunk * 4], srcK + chunk * 4);         // async K -> LDS raw
      vreg[i] = *(const float4*)(srcV + chunk * 4);       // V -> registers
    }
  };

  auto stage_commit = [&](int buf) {
    stage_wait();  // this thread's async K chunks have landed in LDS
    __bf16* kb = sKb[buf];
    __bf16* vt = sVT[buf];
#pragma unroll
    for (int i = 0; i < 4; ++i) {
      const int base = (tid + i * 256) * 4;     // element index, 4 contiguous
      const int key = base >> 6, d = base & 63; // same key for all 4 elems
      const float4 kf = *(const float4*)&sKf[base];
      kb[key * PITCH + d + 0] = (__bf16)kf.x;
      kb[key * PITCH + d + 1] = (__bf16)kf.y;
      kb[key * PITCH + d + 2] = (__bf16)kf.z;
      kb[key * PITCH + d + 3] = (__bf16)kf.w;
      vt[(d + 0) * PITCH + key] = (__bf16)vreg[i].x;      // V transposed
      vt[(d + 1) * PITCH + key] = (__bf16)vreg[i].y;
      vt[(d + 2) * PITCH + key] = (__bf16)vreg[i].z;
      vt[(d + 3) * PITCH + key] = (__bf16)vreg[i].w;
    }
  };

  // ---- running softmax state: row = r + 8*half, replicated over 16-lane half
  float mrow[8], lrow[8];
  v8f   oacc[4];
#pragma unroll
  for (int r = 0; r < 8; ++r) { mrow[r] = -3.0e38f; lrow[r] = 0.0f; }
#pragma unroll
  for (int t = 0; t < 4; ++t) oacc[t] = zero_v8();

  // ---- one key tile; MASKED instantiated at compile time ------------------
  auto tile_body = [&](int kt, auto maskedC) {
    constexpr bool MASKED = decltype(maskedC)::value;
    const int  cur      = kt & 1;
    const bool havenext = (kt + 1 < ktiles);
    if (havenext) stage_issue(kt + 1);   // overlap next tile with compute

    // ---- S = Q * K^T   (4 N-tiles of 16 keys, contraction 64 = 2 x 32)
    v8f sc[4];
#pragma unroll
    for (int nt = 0; nt < 4; ++nt) {
      v8f c = zero_v8();
#pragma unroll
      for (int ch = 0; ch < 2; ++ch) {
        // B-frag: lane (n + 16h) holds K[key = nt*16+n][d = 32ch+8h .. +7, 32ch+16+8h .. +7]
        const uint4* p0 = (const uint4*)&sKb[cur][(nt * 16 + m16) * PITCH + 32 * ch + 8 * half];
        const uint4* p1 = (const uint4*)&sKb[cur][(nt * 16 + m16) * PITCH + 32 * ch + 16 + 8 * half];
        v16bf bf = make_frag(*p0, *p1);
        c = __builtin_amdgcn_wmma_f32_16x16x32_bf16(false, qa[ch], false, bf,
                                                    (short)0, c, false, false);
      }
      sc[nt] = c;
    }

    // ---- online softmax update (masking only in the peeled boundary tile)
    float mnew[8];
#pragma unroll
    for (int r = 0; r < 8; ++r) {
      float mx = mrow[r];
      if constexpr (MASKED) {
#pragma unroll
        for (int nt = 0; nt < 4; ++nt) {
          const int col = kt * KT + nt * 16 + m16;
          float s = sc[nt][r];
          s = (col < valid) ? s : NEGV;
          sc[nt][r] = s;
          mx = fmaxf(mx, s);
        }
      } else {
#pragma unroll
        for (int nt = 0; nt < 4; ++nt) mx = fmaxf(mx, sc[nt][r]);
      }
      mnew[r] = rowmax16(mx);
    }
#pragma unroll
    for (int r = 0; r < 8; ++r) {
      float sum = 0.0f;
#pragma unroll
      for (int nt = 0; nt < 4; ++nt) {
        const float p = __expf(sc[nt][r] - mnew[r]);
        sc[nt][r] = p;
        sum += p;
      }
      sum = rowsum16(sum);
      const float alpha = __expf(mrow[r] - mnew[r]);
      lrow[r] = lrow[r] * alpha + sum;
      mrow[r] = mnew[r];
#pragma unroll
      for (int t = 0; t < 4; ++t) oacc[t][r] *= alpha;
    }

    // ---- P (C-layout) -> wave-private LDS (row-major bf16) -> A-layout frags
    __bf16* pW = &sP[wave * 16 * PITCH];
#pragma unroll
    for (int nt = 0; nt < 4; ++nt)
#pragma unroll
      for (int r = 0; r < 8; ++r)
        pW[(r + 8 * half) * PITCH + nt * 16 + m16] = (__bf16)sc[nt][r];
    asm volatile("s_wait_dscnt 0x0" ::: "memory");  // wave-local LDS exchange

    v16bf pa[2];
#pragma unroll
    for (int c = 0; c < 2; ++c) {
      const uint4* p0 = (const uint4*)&pW[m16 * PITCH + 32 * c + 8 * half];
      const uint4* p1 = (const uint4*)&pW[m16 * PITCH + 32 * c + 16 + 8 * half];
      pa[c] = make_frag(*p0, *p1);
    }

    // ---- O += P * V   (4 d-tiles of 16, contraction 64 keys = 2 x 32)
#pragma unroll
    for (int dt = 0; dt < 4; ++dt) {
#pragma unroll
      for (int ch = 0; ch < 2; ++ch) {
        // B-frag: lane (n + 16h) holds V[key = 32ch+8h .. +7, 32ch+16+8h .. +7][d = dt*16+n]
        const uint4* p0 = (const uint4*)&sVT[cur][(dt * 16 + m16) * PITCH + 32 * ch + 8 * half];
        const uint4* p1 = (const uint4*)&sVT[cur][(dt * 16 + m16) * PITCH + 32 * ch + 16 + 8 * half];
        v16bf bf = make_frag(*p0, *p1);
        oacc[dt] = __builtin_amdgcn_wmma_f32_16x16x32_bf16(false, pa[ch], false, bf,
                                                           (short)0, oacc[dt], false, false);
      }
    }

    // ---- commit next tile into the other buffer, then block-wide sync
    if (havenext) stage_commit((kt + 1) & 1);
    __syncthreads();
  };

  // ---- prologue: stage tile 0, then run full tiles + one peeled masked tile
  stage_issue(0);
  stage_commit(0);
  __syncthreads();

  for (int kt = 0; kt < nfull; ++kt) tile_body(kt, std::false_type{});
  if (haspartial) tile_body(nfull, std::true_type{});

  // ---- epilogue: normalize and store (element row r+8*half, col dt*16+m16)
  float* orow = Out + ((size_t)(b * SEQ + qbase)) * DIM;
  float inv[8];
#pragma unroll
  for (int r = 0; r < 8; ++r) inv[r] = 1.0f / lrow[r];
#pragma unroll
  for (int dt = 0; dt < 4; ++dt)
#pragma unroll
    for (int r = 0; r < 8; ++r)
      orow[(size_t)(r + 8 * half) * DIM + dt * 16 + m16] = oacc[dt][r] * inv[r];
}

// ---------------------------------------------------------------------------
extern "C" void kernel_launch(void* const* d_in, const int* in_sizes, int n_in,
                              void* d_out, int out_size, void* d_ws, size_t ws_size,
                              hipStream_t stream) {
  const float* Q  = (const float*)d_in[0];
  const float* K  = (const float*)d_in[1];
  const float* V  = (const float*)d_in[2];
  const int*   VL = (const int*)d_in[3];
  float* Out = (float*)d_out;

  dim3 grid(BATCH * (SEQ / QT));   // 64 * 8 = 512 blocks
  dim3 block(256);                 // 8 waves (wave32)
  DotProductAttention_flash_bf16_wmma<<<grid, block, 0, stream>>>(Q, K, V, VL, Out);
}
